// HarmonicRNN_53970559041536
// MI455X (gfx1250) — compile-verified, matrix operand order checked by reading, not run on hardware
//
#include <hip/hip_runtime.h>
#include <hip/hip_bf16.h>
#include <math.h>
#include <stdint.h>

#define HIDDEN   512
#define G3       1536     // 3*HIDDEN
#define SRATE    16000
#define TSTEPS   32
#define MAX_MSGS 512
#define NWG      16       // workgroups in the recurrence phase (1 per WGP)
#define JSLICE   32       // hidden units per WG  (NWG*JSLICE == HIDDEN)
#define RROWS    96       // gate rows per WG     (3*JSLICE)

typedef __attribute__((ext_vector_type(16))) __bf16 v16bf;
typedef __attribute__((ext_vector_type(8)))  float  v8f;
typedef int v4i_b128 __attribute__((vector_size(16)));   // matches builtin param

// ---- CDNA5 async global->LDS copy (ASYNCcnt path), with safe fallback -----
#if defined(__has_builtin)
#if __has_builtin(__builtin_amdgcn_global_load_async_to_lds_b128)
#define HAVE_ASYNC_LDS 1
#endif
#endif

__device__ __forceinline__ void copy_f4_to_lds(float* lds_dst, const float* gsrc) {
#if defined(HAVE_ASYNC_LDS)
  // LDS[dst] = MEM[src], 16B per lane, no VGPR staging; tracked by ASYNCcnt.
  // Signature (from hipcc diagnostic): (v4i AS1* src, v4i AS3* dst, Ii, Ii).
  // Low 32 bits of a flat LDS-aperture address are the LDS byte offset.
  __builtin_amdgcn_global_load_async_to_lds_b128(
      (__attribute__((address_space(1))) v4i_b128*)(uintptr_t)gsrc,
      (__attribute__((address_space(3))) v4i_b128*)(uint32_t)(uintptr_t)lds_dst,
      0, 0);
#else
  *(float4*)lds_dst = *(const float4*)gsrc;
#endif
}

__device__ __forceinline__ void wait_async_lds() {
#if defined(__has_builtin)
#if __has_builtin(__builtin_amdgcn_s_wait_asynccnt)
  __builtin_amdgcn_s_wait_asynccnt(0);
#elif defined(__AMDGCN__)
  asm volatile("s_wait_asynccnt 0" ::: "memory");
#endif
#elif defined(__AMDGCN__)
  asm volatile("s_wait_asynccnt 0" ::: "memory");
#endif
}

// ---------------------------------------------------------------------------
// Init: zero midi output, h ping-pong buffers, device barrier counter.
// Runs every launch (deterministic; harness poisons d_out/d_ws only once).
// ---------------------------------------------------------------------------
__global__ void init_k(float* __restrict__ out, float* __restrict__ H,
                       int* __restrict__ bar) {
  const int i = blockIdx.x * blockDim.x + threadIdx.x;
  if (i < MAX_MSGS * 6) out[i] = 0.f;
  if (i < 2 * HIDDEN)   H[i]   = 0.f;
  if (i == 0)           *bar   = 0;
}

// ---------------------------------------------------------------------------
// Phase A: GI[t][1536] = enc_w_ih @ x_t + enc_b_ih for all 32 timesteps.
// GEMM [1536 x 16000] * [16000 x 32] with bf16 WMMA + 2-term bf16 split
// (3 wmma per tile: hi*hi + hi*lo + lo*hi) for ~fp32 accuracy.
// One wave per 16-row M-tile, both N-tiles per wave -> the 98 MB weight is
// streamed from HBM exactly once. Exactly 96 waves: EXEC all-ones per WMMA.
// ---------------------------------------------------------------------------
__global__ __launch_bounds__(128) void enc_in_gemm(
    const float* __restrict__ W,    // [1536][16000] row-major
    const float* __restrict__ X,    // [32][16000]   row-major
    const float* __restrict__ b_ih, // [1536]
    float* __restrict__ GI)         // [32][1536]
{
  const int wave = (blockIdx.x * blockDim.x + threadIdx.x) >> 5; // 0..95
  const int lane = threadIdx.x & 31;
  const int lo16 = lane & 15;
  const int hi   = lane >> 4;      // K-subset selector for A and B
  const int m0   = wave * 16;

  const float* wrow = W + (size_t)(m0 + lo16) * SRATE;   // A: lane -> row m
  const float* x0   = X + (size_t)lo16 * SRATE;          // B ntile0: lane -> t
  const float* x1   = X + (size_t)(16 + lo16) * SRATE;   // B ntile1

  v8f acc0 = {}; v8f acc1 = {};

  for (int k0 = 0; k0 < SRATE; k0 += 32) {
    const int base = k0 + 8 * hi;   // elems 0..7 at base, 8..15 at base+16
    float af[16], b0f[16], b1f[16];
#pragma unroll
    for (int e = 0; e < 8; ++e) {
      af[e]      = wrow[base + e];
      af[e + 8]  = wrow[base + 16 + e];
      b0f[e]     = x0[base + e];
      b0f[e + 8] = x0[base + 16 + e];
      b1f[e]     = x1[base + e];
      b1f[e + 8] = x1[base + 16 + e];
    }
    v16bf ah, al, b0h, b0l, b1h, b1l;
#pragma unroll
    for (int e = 0; e < 16; ++e) {
      __bf16 h  = (__bf16)af[e];  ah[e]  = h;  al[e]  = (__bf16)(af[e]  - (float)h);
      __bf16 h0 = (__bf16)b0f[e]; b0h[e] = h0; b0l[e] = (__bf16)(b0f[e] - (float)h0);
      __bf16 h1 = (__bf16)b1f[e]; b1h[e] = h1; b1l[e] = (__bf16)(b1f[e] - (float)h1);
    }
    acc0 = __builtin_amdgcn_wmma_f32_16x16x32_bf16(false, ah, false, b0h, (short)0, acc0, false, false);
    acc0 = __builtin_amdgcn_wmma_f32_16x16x32_bf16(false, ah, false, b0l, (short)0, acc0, false, false);
    acc0 = __builtin_amdgcn_wmma_f32_16x16x32_bf16(false, al, false, b0h, (short)0, acc0, false, false);
    acc1 = __builtin_amdgcn_wmma_f32_16x16x32_bf16(false, ah, false, b1h, (short)0, acc1, false, false);
    acc1 = __builtin_amdgcn_wmma_f32_16x16x32_bf16(false, ah, false, b1l, (short)0, acc1, false, false);
    acc1 = __builtin_amdgcn_wmma_f32_16x16x32_bf16(false, al, false, b1h, (short)0, acc1, false, false);
  }

  // C layout: elem g -> M = m0 + g + 8*hi ; N(t) = lo16 (+16 for ntile1)
#pragma unroll
  for (int g = 0; g < 8; ++g) {
    const int mm = m0 + g + 8 * hi;
    const float bias = b_ih[mm];
    GI[(size_t)lo16 * G3 + mm]        = acc0[g] + bias;
    GI[(size_t)(16 + lo16) * G3 + mm] = acc1[g] + bias;
  }
}

// ---------------------------------------------------------------------------
// Phase B: sequential recurrence across NWG=16 WGPs. Each WG keeps its
// 96x512 f32 slice of w_hh resident in LDS (192 KB -- CDNA5's 320 KB/WGP),
// preloaded with async global->LDS b128 copies. Each step moves only the
// 2 KB hidden vector through L2 plus one device-scope barrier. h is
// ping-ponged in global so one barrier/step suffices. All WGs recompute msg
// bit-identically -> uniform early exit (same barrier count everywhere).
// ---------------------------------------------------------------------------
__global__ __launch_bounds__(256) void rnn_seq(
    const float* __restrict__ GI,       // [32][1536] (enc b_ih folded in)
    const float* __restrict__ enc_w_hh, // [1536][512] row-major
    const float* __restrict__ enc_b_hh, // [1536]
    const float* __restrict__ dec_w_ih, // [1536][6]
    const float* __restrict__ dec_b_ih, // [1536]
    const float* __restrict__ dec_w_hh, // [1536][512]
    const float* __restrict__ dec_b_hh, // [1536]
    const float* __restrict__ lin_w,    // [6][512]
    const float* __restrict__ lin_b,    // [6]
    float* __restrict__ Hbuf,           // [2][512] ping-pong (zeroed by init)
    int*   __restrict__ bar,            // barrier counter (zeroed by init)
    float* __restrict__ out)            // 3072 midi floats + 1 int slot
{
  __shared__ __align__(16) float Wlds[RROWS * HIDDEN];   // 192 KB weight slice
  __shared__ __align__(16) float linw_s[6 * HIDDEN];     // 12 KB
  __shared__ float bias_s[RROWS];          // b_hh slice
  __shared__ float dih_s[RROWS * 6];       // dec w_ih slice
  __shared__ float dbi_s[RROWS];           // dec b_ih slice
  __shared__ float gh_s[RROWS];
  __shared__ float last_s[8];
  __shared__ float msg_s[8];
  __shared__ int   msgcount;

  const int tid  = threadIdx.x;
  const int g    = blockIdx.x;            // WG id: owns hidden units [g*32, g*32+32)
  const bool wg0 = (g == 0);
  const int wv   = tid >> 5;
  const int lane = tid & 31;

  float* Hp = Hbuf;            // read buffer (current h)
  float* Hq = Hbuf + HIDDEN;   // write buffer (next h)
  int gen = 0;

  // local row lr -> global gate row: (lr/32)*512 + g*32 + (lr%32)
  auto growof = [&](int lr) { return (lr >> 5) * HIDDEN + g * JSLICE + (lr & 31); };

  auto load_slice = [&](const float* Whh, const float* bhh) {
    // 96 rows x 512 f32 = 12288 float4 async copies, 48 per thread
    for (int q = tid; q < RROWS * (HIDDEN / 4); q += 256) {
      const int lr = q >> 7;                 // q / 128
      const int k4 = (q & 127) << 2;         // 4-float column offset
      copy_f4_to_lds(Wlds + (size_t)lr * HIDDEN + k4,
                     Whh + (size_t)growof(lr) * HIDDEN + k4);
    }
    for (int lr = tid; lr < RROWS; lr += 256) bias_s[lr] = bhh[growof(lr)];
  };

  auto gbarrier = [&]() {
    ++gen;
    __syncthreads();
    if (tid == 0) {
      __threadfence();  // release this WG's global stores
      __hip_atomic_fetch_add(bar, 1, __ATOMIC_ACQ_REL, __HIP_MEMORY_SCOPE_AGENT);
      while (__hip_atomic_load(bar, __ATOMIC_ACQUIRE, __HIP_MEMORY_SCOPE_AGENT)
             < NWG * gen)
        __builtin_amdgcn_s_sleep(1);
    }
    __syncthreads();
    __threadfence();  // acquire: all threads see other WGs' stores
  };

  // gh = Wlds @ h + bias : 8 waves x 12 rows, lane-split K with shfl reduce
  auto matvec = [&](const float* __restrict__ Hr) {
    const float4 h0 = *(const float4*)(Hr + 4 * lane);
    const float4 h1 = *(const float4*)(Hr + 4 * lane + 128);
    const float4 h2 = *(const float4*)(Hr + 4 * lane + 256);
    const float4 h3 = *(const float4*)(Hr + 4 * lane + 384);
#pragma unroll 4
    for (int rr = 0; rr < 12; ++rr) {
      const int lr = wv * 12 + rr;
      const float* wr = Wlds + (size_t)lr * HIDDEN;
      const float4 w0 = *(const float4*)(wr + 4 * lane);
      const float4 w1 = *(const float4*)(wr + 4 * lane + 128);
      const float4 w2 = *(const float4*)(wr + 4 * lane + 256);
      const float4 w3 = *(const float4*)(wr + 4 * lane + 384);
      float p = 0.f;
      p = fmaf(w0.x, h0.x, p); p = fmaf(w0.y, h0.y, p);
      p = fmaf(w0.z, h0.z, p); p = fmaf(w0.w, h0.w, p);
      p = fmaf(w1.x, h1.x, p); p = fmaf(w1.y, h1.y, p);
      p = fmaf(w1.z, h1.z, p); p = fmaf(w1.w, h1.w, p);
      p = fmaf(w2.x, h2.x, p); p = fmaf(w2.y, h2.y, p);
      p = fmaf(w2.z, h2.z, p); p = fmaf(w2.w, h2.w, p);
      p = fmaf(w3.x, h3.x, p); p = fmaf(w3.y, h3.y, p);
      p = fmaf(w3.w, h3.w, p); p = fmaf(w3.z, h3.z, p);
      for (int off = 16; off > 0; off >>= 1) p += __shfl_down(p, off);
      if (lane == 0) gh_s[lr] = p + bias_s[lr];
    }
    __syncthreads();
  };

  auto sigm = [](float v) { return 1.f / (1.f + expf(-v)); };

  // msg = lin_w @ h_new + lin_b, 6 rows by waves 0..5, redundant in every WG
  auto do_msg = [&](const float* __restrict__ Hn) {
    if (tid < 192) {
      const int j = wv;
      float p = 0.f;
#pragma unroll 4
      for (int i = 0; i < 16; ++i) {
        const int k = lane + 32 * i;
        p = fmaf(linw_s[j * HIDDEN + k], Hn[k], p);
      }
      for (int off = 16; off > 0; off >>= 1) p += __shfl_down(p, off);
      if (lane == 0) msg_s[j] = p + lin_b[j];
    }
    __syncthreads();
  };

  // ---- prologue: encoder weight slice + lin_w into LDS --------------------
  load_slice(enc_w_hh, enc_b_hh);
  for (int i = tid; i < 6 * (HIDDEN / 4); i += 256)
    copy_f4_to_lds(linw_s + 4 * i, lin_w + 4 * i);
  if (tid == 0) msgcount = 1;
  wait_async_lds();
  __syncthreads();

  // ---- encoder: 32 steps --------------------------------------------------
  for (int t = 0; t < TSTEPS; ++t) {
    matvec(Hp);
    if (tid < JSLICE) {
      const int j = g * JSLICE + tid;
      const float* gi = GI + (size_t)t * G3;
      const float r = sigm(gi[j] + gh_s[tid]);
      const float z = sigm(gi[HIDDEN + j] + gh_s[JSLICE + tid]);
      const float n = tanhf(gi[2 * HIDDEN + j] + r * gh_s[2 * JSLICE + tid]);
      Hq[j] = (1.f - z) * n + z * Hp[j];
    }
    gbarrier();
    float* tmp = Hp; Hp = Hq; Hq = tmp;
  }

  // ---- switch LDS slice to decoder weights --------------------------------
  load_slice(dec_w_hh, dec_b_hh);
  for (int i = tid; i < RROWS * 6; i += 256)
    dih_s[i] = dec_w_ih[(size_t)growof(i / 6) * 6 + (i % 6)];
  for (int lr = tid; lr < RROWS; lr += 256) dbi_s[lr] = dec_b_ih[growof(lr)];
  if (tid < 6) last_s[tid] = 0.f;
  wait_async_lds();
  __syncthreads();

  auto dec_combine = [&]() {
    if (tid < JSLICE) {
      const int j = g * JSLICE + tid;
      float gr = dbi_s[tid], gz = dbi_s[JSLICE + tid], gn = dbi_s[2 * JSLICE + tid];
#pragma unroll
      for (int k = 0; k < 6; ++k) {
        const float lv = last_s[k];
        gr = fmaf(dih_s[tid * 6 + k], lv, gr);
        gz = fmaf(dih_s[(JSLICE + tid) * 6 + k], lv, gz);
        gn = fmaf(dih_s[(2 * JSLICE + tid) * 6 + k], lv, gn);
      }
      const float r = sigm(gr + gh_s[tid]);
      const float z = sigm(gz + gh_s[JSLICE + tid]);
      const float n = tanhf(gn + r * gh_s[2 * JSLICE + tid]);
      Hq[j] = (1.f - z) * n + z * Hp[j];
    }
  };

  // ---- decoder step 0 (zero input) ---------------------------------------
  matvec(Hp);
  dec_combine();
  gbarrier();
  do_msg(Hq);
  if (wg0 && tid < 6) out[tid] = msg_s[tid];
  if (tid < 6) last_s[tid] = msg_s[tid];
  __syncthreads();
  { float* tmp = Hp; Hp = Hq; Hq = tmp; }

  // ---- decoder loop (uniform early exit == reference's latched cond) ------
  for (int it = 0; it < MAX_MSGS - 2; ++it) {
    const bool cond = (rintf(last_s[0]) != 2.0f) && (msgcount < MAX_MSGS - 1);
    if (!cond) break;   // identical in every thread of every WG
    matvec(Hp);
    dec_combine();
    gbarrier();
    do_msg(Hq);
    if (tid == 0) msgcount++;
    __syncthreads();
    if (wg0 && tid < 6) out[(size_t)msgcount * 6 + tid] = msg_s[tid];
    if (tid < 6) last_s[tid] = msg_s[tid];
    __syncthreads();
    float* tmp = Hp; Hp = Hq; Hq = tmp;
  }
  if (wg0 && tid == 0)
    reinterpret_cast<int*>(out)[MAX_MSGS * 6] = msgcount + 1;  // num_messages
}

// ---------------------------------------------------------------------------
extern "C" void kernel_launch(void* const* d_in, const int* in_sizes, int n_in,
                              void* d_out, int out_size, void* d_ws, size_t ws_size,
                              hipStream_t stream) {
  const float* x        = (const float*)d_in[0];   // [1][32][16000]
  const float* enc_w_ih = (const float*)d_in[1];   // [1536][16000]
  const float* enc_w_hh = (const float*)d_in[2];   // [1536][512]
  const float* enc_b_ih = (const float*)d_in[3];   // [1536]
  const float* enc_b_hh = (const float*)d_in[4];   // [1536]
  const float* dec_w_ih = (const float*)d_in[5];   // [1536][6]
  const float* dec_w_hh = (const float*)d_in[6];   // [1536][512]
  const float* dec_b_ih = (const float*)d_in[7];   // [1536]
  const float* dec_b_hh = (const float*)d_in[8];   // [1536]
  const float* lin_w    = (const float*)d_in[9];   // [6][512]
  const float* lin_b    = (const float*)d_in[10];  // [6]
  float* out = (float*)d_out;

  // workspace: GI | H ping-pong | barrier counter
  float* GI   = (float*)d_ws;             // 32*1536 floats
  float* Hbuf = GI + TSTEPS * G3;         // 2*512 floats
  int*   bar  = (int*)(Hbuf + 2 * HIDDEN);

  init_k<<<(MAX_MSGS * 6 + 255) / 256, 256, 0, stream>>>(out, Hbuf, bar);

  // Phase A: WMMA bf16-split GEMM, exactly 96 waves (EXEC all-ones).
  enc_in_gemm<<<24, 128, 0, stream>>>(enc_w_ih, x, enc_b_ih, GI);

  // Phase B: 16 persistent WGs (1/WGP), LDS-resident weight slices.
  rnn_seq<<<NWG, 256, 0, stream>>>(GI, enc_w_hh, enc_b_hh,
                                   dec_w_ih, dec_b_ih, dec_w_hh, dec_b_hh,
                                   lin_w, lin_b, Hbuf, bar, out);
}
